// AttentionLayer_74131135529982
// MI455X (gfx1250) — matrix operand order repaired and checked by reading, not compiled, not run on hardware
//
#include <hip/hip_runtime.h>
#include <hip/hip_bf16.h>
#include <stdint.h>

// ---------------------------------------------------------------- types
typedef __attribute__((ext_vector_type(16))) __bf16 v16bf;
typedef __attribute__((ext_vector_type(8)))  __bf16 v8bf;
typedef __attribute__((ext_vector_type(8)))  float  v8f;

#define S_   2048
#define D_   4096
#define H_   32
#define KVH_ 8
#define HD_  128

__device__ __forceinline__ v8f wmma_bf16(v16bf a, v16bf b, v8f c) {
  // D = A(16x32 bf16) * B(32x16 bf16) + C(16x16 f32)
  return __builtin_amdgcn_wmma_f32_16x16x32_bf16(false, a, false, b, (short)0, c, false, false);
}

// CDNA5 async DMA: global -> LDS, 16 bytes per lane, tracked by ASYNCcnt.
// LDS destination is the low 32 bits of the generic pointer (LDS aperture).
__device__ __forceinline__ void async_load_b128(void* lds_dst, const void* gsrc) {
  uint32_t lds_off = (uint32_t)(uintptr_t)lds_dst;
  asm volatile("global_load_async_to_lds_b128 %0, %1, off"
               :: "v"(lds_off), "v"(gsrc) : "memory");
}
__device__ __forceinline__ void wait_asynccnt0() {
  asm volatile("s_wait_asynccnt 0" ::: "memory");
}

// A fragment (16x32, M=row): lane<16 -> row=lane, K {0..7,16..23}; lane>=16 -> K {8..15,24..31}
__device__ __forceinline__ v16bf load_a_frag(const __bf16* base, int ld) {
  const int lane = threadIdx.x & 31;
  const int half = lane >> 4, m = lane & 15;
  const __bf16* p = base + m * ld + half * 8;
  v16bf r;
  *((v8bf*)&r)     = *(const v8bf*)p;
  *((v8bf*)&r + 1) = *(const v8bf*)(p + 16);
  return r;
}

// B fragment (32x16, N=col): lane<16 -> col=lane, K {0..7,16..23}; lane>=16 -> K {8..15,24..31}
__device__ __forceinline__ v16bf load_b_frag(const __bf16* base, int ld) {
  const int lane = threadIdx.x & 31;
  const int half = lane >> 4, n = lane & 15;
  const __bf16* p = base + n * ld + half * 8;
  v16bf r;
  *((v8bf*)&r)     = *(const v8bf*)p;
  *((v8bf*)&r + 1) = *(const v8bf*)(p + 16);
  return r;
}

__device__ __forceinline__ float half16_max(float v) {
  #pragma unroll
  for (int m = 1; m < 16; m <<= 1) v = fmaxf(v, __shfl_xor(v, m, 32));
  return v;
}
__device__ __forceinline__ float half16_sum(float v) {
  #pragma unroll
  for (int m = 1; m < 16; m <<= 1) v += __shfl_xor(v, m, 32);
  return v;
}

// ---------------------------------------------------------------- f32 -> bf16
__global__ __launch_bounds__(256) void cvt_bf16_kernel(const float* __restrict__ in,
                                                       __bf16* __restrict__ out, int n) {
  int i = blockIdx.x * 256 + threadIdx.x;
  if (i < n) out[i] = (__bf16)in[i];
}

// ---------------------------------------------------------------- RMSNorm -> bf16
__global__ __launch_bounds__(256) void rmsnorm_kernel(const float* __restrict__ x,
                                                      const float* __restrict__ w,
                                                      __bf16* __restrict__ xn) {
  __shared__ float red[8];
  __shared__ float stotal;
  const int row = blockIdx.x;
  const float* xr = x + (size_t)row * D_;
  float ss = 0.f;
  for (int i = threadIdx.x; i < D_; i += 256) { float v = xr[i]; ss += v * v; }
  #pragma unroll
  for (int m = 1; m < 32; m <<= 1) ss += __shfl_xor(ss, m, 32);
  if ((threadIdx.x & 31) == 0) red[threadIdx.x >> 5] = ss;
  __syncthreads();
  if (threadIdx.x == 0) {
    float t = 0.f;
    #pragma unroll
    for (int i = 0; i < 8; ++i) t += red[i];
    stotal = t;
  }
  __syncthreads();
  const float scale = rsqrtf(stotal / (float)D_ + 1e-5f);
  for (int i = threadIdx.x; i < D_; i += 256)
    xn[(size_t)row * D_ + i] = (__bf16)(xr[i] * scale * w[i]);
}

// ---------------------------------------------------------------- RoPE f32 -> bf16
__global__ __launch_bounds__(256) void rope_kernel(const float* __restrict__ t,
                                                   const float* __restrict__ freqs,
                                                   __bf16* __restrict__ out,
                                                   int heads, int total_pairs) {
  int p = blockIdx.x * 256 + threadIdx.x;
  if (p >= total_pairs) return;
  const int j  = p & 63;        // HD/2 = 64
  const int sh = p >> 6;        // s*heads + h
  const int s  = sh / heads;
  float sn, cs;
  __sincosf((float)s * freqs[j], &sn, &cs);
  const size_t base = (size_t)sh * HD_;
  const float re = t[base + 2 * j], im = t[base + 2 * j + 1];
  out[base + 2 * j]     = (__bf16)(re * cs - im * sn);
  out[base + 2 * j + 1] = (__bf16)(re * sn + im * cs);
}

// ---------------------------------------------------------------- GEMM: C[MxN] = A[MxK] * B[NxK]^T
// A,B bf16 row-major (contraction along K); C f32. Block = 128x128 tile, 8 waves.
// Double-buffered async global->LDS staging; sched_group_barrier pins
// "all DS reads, then all WMMAs" so waitcnt can issue partial DS waits.
__global__ __launch_bounds__(256) void gemm_bf16_kernel(const __bf16* __restrict__ A,
                                                        const __bf16* __restrict__ B,
                                                        float* __restrict__ C,
                                                        int M, int N, int K) {
  __shared__ __bf16 As[2][128 * 32];
  __shared__ __bf16 Bs[2][128 * 32];
  const int tid = threadIdx.x;
  const int lane = tid & 31, wave = tid >> 5;
  const int half = lane >> 4, l16 = lane & 15;
  const int m0 = blockIdx.x * 128, n0 = blockIdx.y * 128;

  v8f acc[8] = {};
  const int r = tid >> 1, c = (tid & 1) * 16;  // tile-load assignment (32B per thread)

  auto issue_tile = [&](int buf, int k0) {
    const __bf16* srcA = A + (size_t)(m0 + r) * K + k0 + c;
    const __bf16* srcB = B + (size_t)(n0 + r) * K + k0 + c;
    async_load_b128(&As[buf][r * 32 + c],     srcA);
    async_load_b128(&As[buf][r * 32 + c + 8], srcA + 8);
    async_load_b128(&Bs[buf][r * 32 + c],     srcB);
    async_load_b128(&Bs[buf][r * 32 + c + 8], srcB + 8);
  };

  issue_tile(0, 0);
  wait_asynccnt0();
  __syncthreads();

  int buf = 0;
  for (int k0 = 0; k0 < K; k0 += 32) {
    if (k0 + 32 < K) issue_tile(buf ^ 1, k0 + 32);     // overlap next tile DMA with math
    const v16bf af = load_a_frag(&As[buf][wave * 16 * 32], 32);
    v16bf bfr[8];
    #pragma unroll
    for (int j = 0; j < 8; ++j) bfr[j] = load_b_frag(&Bs[buf][j * 16 * 32], 32);
    #pragma unroll
    for (int j = 0; j < 8; ++j) acc[j] = wmma_bf16(af, bfr[j], acc[j]);
    // Schedule shape: 18 DS reads (A frag + 8 B frags) first, then 8 WMMAs.
    __builtin_amdgcn_sched_group_barrier(0x100, 18, 0);  // DS read
    __builtin_amdgcn_sched_group_barrier(0x008,  8, 0);  // MFMA/WMMA
    wait_asynccnt0();
    __syncthreads();
    buf ^= 1;
  }

  // C tile store: lane=col (per half), VGPR index = row (lane>=16 -> row+8)
  #pragma unroll
  for (int j = 0; j < 8; ++j)
    #pragma unroll
    for (int rr = 0; rr < 8; ++rr) {
      const int row = m0 + wave * 16 + rr + half * 8;
      const int col = n0 + j * 16 + l16;
      C[(size_t)row * N + col] = acc[j][rr];
    }
}

// ---------------------------------------------------------------- Flash attention (causal, GQA)
// Block: 128 query rows x 1 head. 8 waves, 16 query rows each. KV step = 32.
__global__ __launch_bounds__(256) void attn_kernel(const __bf16* __restrict__ Q,
                                                   const __bf16* __restrict__ Kt,
                                                   const __bf16* __restrict__ Vt,
                                                   __bf16* __restrict__ O) {
  __shared__ __bf16 Ks[32 * 128];    // [key][d]
  __shared__ __bf16 VTs[128 * 32];   // [d][key]
  __shared__ __bf16 Ps[8][16 * 32];  // per-wave P staging
  const int tid = threadIdx.x;
  const int lane = tid & 31, wave = tid >> 5;
  const int half = lane >> 4, l16 = lane & 15;
  const int qbase = blockIdx.x * 128;
  const int h = blockIdx.y;
  const int g = h >> 2;              // H/KVH = 4
  const int qrow0 = qbase + wave * 16;

  // Q A-fragments for the 4 K-chunks of HD=128 (resident for whole kernel)
  v16bf qf[4];
  {
    const __bf16* qp = Q + (size_t)(qrow0 + l16) * (H_ * HD_) + h * HD_ + half * 8;
    #pragma unroll
    for (int kk = 0; kk < 4; ++kk) {
      v16bf rr;
      *((v8bf*)&rr)     = *(const v8bf*)(qp + kk * 32);
      *((v8bf*)&rr + 1) = *(const v8bf*)(qp + kk * 32 + 16);
      qf[kk] = rr;
    }
  }

  v8f acc[8] = {};
  float mr[8], lr[8];
  #pragma unroll
  for (int i = 0; i < 8; ++i) { mr[i] = -1e30f; lr[i] = 0.f; }

  const float sc = 0.08838834764831845f;  // 1/sqrt(128)
  const int kv_end = qbase + 128;

  for (int kv0 = 0; kv0 < kv_end; kv0 += 32) {
    __syncthreads();
    {  // K tile via async DMA; V tile transposed through VGPRs
      const int r = tid >> 3, c = (tid & 7) * 16;
      const __bf16* ksrc = Kt + (size_t)(kv0 + r) * (KVH_ * HD_) + g * HD_ + c;
      async_load_b128(&Ks[r * 128 + c],     ksrc);
      async_load_b128(&Ks[r * 128 + c + 8], ksrc + 8);
      const __bf16* vsrc = Vt + (size_t)(kv0 + r) * (KVH_ * HD_) + g * HD_ + c;
      v8bf t0 = *(const v8bf*)vsrc;
      v8bf t1 = *(const v8bf*)(vsrc + 8);
      #pragma unroll
      for (int i = 0; i < 8; ++i) {
        VTs[(c + i) * 32 + r]     = t0[i];
        VTs[(c + 8 + i) * 32 + r] = t1[i];
      }
    }
    wait_asynccnt0();
    __syncthreads();

    // S = Q * K^T  (two 16-key column tiles); preload B-frags, then WMMA chain
    v8f s0 = {}, s1 = {};
    {
      v16bf kb[8];
      #pragma unroll
      for (int kk = 0; kk < 4; ++kk) {
        kb[kk]     = load_b_frag(&Ks[0]        + kk * 32, 128);
        kb[4 + kk] = load_b_frag(&Ks[16 * 128] + kk * 32, 128);
      }
      #pragma unroll
      for (int kk = 0; kk < 4; ++kk) {
        s0 = wmma_bf16(qf[kk], kb[kk],     s0);
        s1 = wmma_bf16(qf[kk], kb[4 + kk], s1);
      }
      __builtin_amdgcn_sched_group_barrier(0x100, 16, 0);  // 16 DS reads
      __builtin_amdgcn_sched_group_barrier(0x008,  8, 0);  // 8 WMMAs
    }
    // scale + causal mask (lane holds key column; VGPR index -> query row)
    #pragma unroll
    for (int rr = 0; rr < 8; ++rr) {
      const int q_idx = qrow0 + rr + half * 8;
      s0[rr] = (kv0 + l16      <= q_idx) ? s0[rr] * sc : -1e30f;
      s1[rr] = (kv0 + 16 + l16 <= q_idx) ? s1[rr] * sc : -1e30f;
    }
    // online softmax, stage P (bf16) into per-wave LDS in A-fragment source order
    #pragma unroll
    for (int rr = 0; rr < 8; ++rr) {
      const float rowmax = half16_max(fmaxf(s0[rr], s1[rr]));
      const float nm = fmaxf(mr[rr], rowmax);
      const float f  = __expf(mr[rr] - nm);
      const float p0 = __expf(s0[rr] - nm);
      const float p1 = __expf(s1[rr] - nm);
      lr[rr] = lr[rr] * f + half16_sum(p0 + p1);
      mr[rr] = nm;
      #pragma unroll
      for (int j = 0; j < 8; ++j) acc[j][rr] *= f;
      const int m = rr + half * 8;
      Ps[wave][m * 32 + l16]      = (__bf16)p0;
      Ps[wave][m * 32 + 16 + l16] = (__bf16)p1;
    }
    __builtin_amdgcn_wave_barrier();
    asm volatile("s_wait_dscnt 0" ::: "memory");  // wave-private staging: DS-count wait only

    // O += P(16x32) * V(32x128); preload V-frags, then WMMA chain
    {
      const v16bf pf = load_a_frag(Ps[wave], 32);
      v16bf vb[8];
      #pragma unroll
      for (int j = 0; j < 8; ++j) vb[j] = load_b_frag(&VTs[j * 16 * 32], 32);
      #pragma unroll
      for (int j = 0; j < 8; ++j) acc[j] = wmma_bf16(pf, vb[j], acc[j]);
      __builtin_amdgcn_sched_group_barrier(0x100, 18, 1);  // 18 DS reads
      __builtin_amdgcn_sched_group_barrier(0x008,  8, 1);  // 8 WMMAs
    }
  }

  // normalize + store bf16
  #pragma unroll
  for (int j = 0; j < 8; ++j)
    #pragma unroll
    for (int rr = 0; rr < 8; ++rr) {
      const int row = qrow0 + rr + half * 8;
      const int col = h * HD_ + j * 16 + l16;
      O[(size_t)row * (H_ * HD_) + col] = (__bf16)(acc[j][rr] / lr[rr]);
    }
}

// ---------------------------------------------------------------- launch
extern "C" void kernel_launch(void* const* d_in, const int* in_sizes, int n_in,
                              void* d_out, int out_size, void* d_ws, size_t ws_size,
                              hipStream_t stream) {
  const float* x  = (const float*)d_in[0];
  const float* nw = (const float*)d_in[1];
  const float* wq = (const float*)d_in[2];
  const float* wk = (const float*)d_in[3];
  const float* wv = (const float*)d_in[4];
  const float* wo = (const float*)d_in[5];
  const float* fr = (const float*)d_in[6];
  float* out = (float*)d_out;

  char* ws = (char*)d_ws;
  size_t off = 0;
  auto alloc = [&](size_t bytes) -> void* {
    void* p = ws + off;
    off += (bytes + 255) & ~(size_t)255;
    return p;
  };
  const size_t NQ = (size_t)H_ * HD_ * D_;    // 16.7M
  const size_t NK = (size_t)KVH_ * HD_ * D_;  // 4.2M
  __bf16* wq_b = (__bf16*)alloc(NQ * 2);
  __bf16* wk_b = (__bf16*)alloc(NK * 2);
  __bf16* wv_b = (__bf16*)alloc(NK * 2);
  __bf16* wo_b = (__bf16*)alloc(NQ * 2);
  __bf16* xn   = (__bf16*)alloc((size_t)S_ * D_ * 2);
  float*  qf   = (float*)alloc((size_t)S_ * H_ * HD_ * 4);
  float*  kf   = (float*)alloc((size_t)S_ * KVH_ * HD_ * 4);
  float*  vf   = (float*)alloc((size_t)S_ * KVH_ * HD_ * 4);
  __bf16* q_b  = (__bf16*)alloc((size_t)S_ * H_ * HD_ * 2);
  __bf16* k_b  = (__bf16*)alloc((size_t)S_ * KVH_ * HD_ * 2);
  __bf16* v_b  = (__bf16*)alloc((size_t)S_ * KVH_ * HD_ * 2);
  __bf16* o_b  = (__bf16*)qf;  // qf dead after RoPE; reuse 32MB region for 16MB o

  // 1. weights -> bf16
  cvt_bf16_kernel<<<(int)((NQ + 255) / 256), 256, 0, stream>>>(wq, wq_b, (int)NQ);
  cvt_bf16_kernel<<<(int)((NK + 255) / 256), 256, 0, stream>>>(wk, wk_b, (int)NK);
  cvt_bf16_kernel<<<(int)((NK + 255) / 256), 256, 0, stream>>>(wv, wv_b, (int)NK);
  cvt_bf16_kernel<<<(int)((NQ + 255) / 256), 256, 0, stream>>>(wo, wo_b, (int)NQ);

  // 2. RMSNorm
  rmsnorm_kernel<<<S_, 256, 0, stream>>>(x, nw, xn);

  // 3. QKV projections (WMMA)
  gemm_bf16_kernel<<<dim3(S_ / 128, (H_ * HD_) / 128), 256, 0, stream>>>(xn, wq_b, qf, S_, H_ * HD_, D_);
  gemm_bf16_kernel<<<dim3(S_ / 128, (KVH_ * HD_) / 128), 256, 0, stream>>>(xn, wk_b, kf, S_, KVH_ * HD_, D_);
  gemm_bf16_kernel<<<dim3(S_ / 128, (KVH_ * HD_) / 128), 256, 0, stream>>>(xn, wv_b, vf, S_, KVH_ * HD_, D_);

  // 4. RoPE (f32 in, bf16 out) + V convert
  {
    const int qp = S_ * H_ * (HD_ / 2), kp = S_ * KVH_ * (HD_ / 2);
    rope_kernel<<<(qp + 255) / 256, 256, 0, stream>>>(qf, fr, q_b, H_, qp);
    rope_kernel<<<(kp + 255) / 256, 256, 0, stream>>>(kf, fr, k_b, KVH_, kp);
    const int nv = S_ * KVH_ * HD_;
    cvt_bf16_kernel<<<(nv + 255) / 256, 256, 0, stream>>>(vf, v_b, nv);
  }

  // 5. Flash attention (WMMA) -> o_b bf16
  attn_kernel<<<dim3(S_ / 128, H_), 256, 0, stream>>>(q_b, k_b, v_b, o_b);

  // 6. Output projection (WMMA) -> f32 out
  gemm_bf16_kernel<<<dim3(S_ / 128, D_ / 128), 256, 0, stream>>>(o_b, wo_b, out, S_, D_, H_ * HD_);

  (void)in_sizes; (void)n_in; (void)out_size; (void)ws_size;
}
